// FloquetSolver_90701119357169
// MI455X (gfx1250) — matrix operand order repaired and checked by reading, not compiled, not run on hardware
//
#include <hip/hip_runtime.h>
#include <hip/hip_bf16.h>

// ---------------- problem constants (from reference) ----------------
#define F_NODE 64
#define F_EDGE 8
#define DIMQ   4
#define BZ     8
#define NODES  (DIMQ * (2 * BZ + 1))   // 68
#define BGRAPH 1024
#define NN     (BGRAPH * NODES)        // 69632
#define EE     (32 * NN)               // 2228224
#define ETILES 8                       // edge tiles (of 128 edges) per block

static_assert(NN % 256 == 0, "encoder grid exact");
static_assert(NN % 128 == 0, "node tile exact");
static_assert(EE % (128 * ETILES) == 0, "edge grid exact");
static_assert((NN * 64) % 1024 == 0, "zero grid exact");

typedef __attribute__((ext_vector_type(16))) _Float16 v16h;
typedef __attribute__((ext_vector_type(8)))  _Float16 v8h;
typedef __attribute__((ext_vector_type(4)))  _Float16 v4h;
typedef __attribute__((ext_vector_type(8)))  float    v8f;

// ---------------- CDNA5 async memory->LDS helpers ----------------
// GLOBAL_LOAD_ASYNC_TO_LDS_B128: per-lane 16B global -> LDS, tracked by ASYNCcnt.
__device__ __forceinline__ void async_gather_b128(unsigned lds_off, const void* gaddr) {
  asm volatile("global_load_async_to_lds_b128 %0, %1, off"
               :: "v"(lds_off), "v"(gaddr) : "memory");
}
__device__ __forceinline__ void wait_asynccnt0() {
  asm volatile("s_wait_asynccnt 0x0" ::: "memory");
}
// Truncate a generic pointer to a shared var to its LDS byte offset
// (flat LDS aperture keeps the offset in addr[31:0]).
__device__ __forceinline__ unsigned lds_offset(const void* p) {
  return (unsigned)(size_t)p;
}

// ---------------- WMMA helpers (CDNA5 16x16x32 f16 -> f32) ----------------
__device__ __forceinline__ v16h cat8(v8h lo, v8h hi) {
  return __builtin_shufflevector(lo, hi, 0,1,2,3,4,5,6,7,8,9,10,11,12,13,14,15);
}

// A fragment: 16x32 f16 tile, row-major in LDS with stride ldin (halves).
// Lane L: row M = L&15; a[0..7] = K(kOff + 8*(L>>4) + j); a[8..15] = K(kOff+16 + 8*(L>>4) + j).
__device__ __forceinline__ v16h load_a_frag(const _Float16* sIn, int ldin,
                                            int mlo, int hi, int kOff) {
  const v8h* p0 = (const v8h*)(sIn + mlo * ldin + kOff + hi * 8);
  const v8h* p1 = (const v8h*)(sIn + mlo * ldin + kOff + 16 + hi * 8);
  return cat8(*p0, *p1);
}

// B fragment: 32x16 f16 tile of W (KxN), stored transposed [n][k] in LDS, stride ldk.
// Lane L: col N = L&15; b[0..7] = W[kOff + 8*(L>>4) + j][col]; b[8..15] = W[kOff+16 + 8*(L>>4) + j][col].
__device__ __forceinline__ v16h load_b_frag(const _Float16* sWt, int ldk,
                                            int mlo, int hi, int kOff, int nOff) {
  const _Float16* q = sWt + (size_t)(nOff + mlo) * ldk + kOff + hi * 8;
  return cat8(*(const v8h*)q, *(const v8h*)(q + 16));
}

__device__ __forceinline__ v8f wmma_f16(v16h a, v16h b, v8f c) {
  // (neg_a, A, neg_b, B, c_mod, C, reuse_a, reuse_b)
  return __builtin_amdgcn_wmma_f32_16x16x32_f16(false, a, false, b, (short)0, c,
                                                false, false);
}

// One GEMM stage for a wave: D(16x64) = relu(A(16x(KT*32)) @ W + bias) -> f16 LDS out (stride 64).
template <int KT>
__device__ __forceinline__ void stage_relu(const _Float16* sIn, int ldin,
                                           const _Float16* sWt,  // [64][KT*32] transposed
                                           const float* sBias,
                                           _Float16* sOut, int lane) {
  const int mlo = lane & 15, hi = lane >> 4;
  const int ldk = KT * 32;
  v16h a[KT];
#pragma unroll
  for (int kt = 0; kt < KT; ++kt) a[kt] = load_a_frag(sIn, ldin, mlo, hi, kt * 32);
#pragma unroll
  for (int nt = 0; nt < 4; ++nt) {
    v8f c = {};
#pragma unroll
    for (int kt = 0; kt < KT; ++kt)
      c = wmma_f16(a[kt], load_b_frag(sWt, ldk, mlo, hi, kt * 32, nt * 16), c);
    const float bn = sBias[nt * 16 + mlo];
#pragma unroll
    for (int r = 0; r < 8; ++r) {
      float v = c[r] + bn;
      sOut[(r + 8 * hi) * 64 + nt * 16 + mlo] = (_Float16)(v > 0.f ? v : 0.f);
    }
  }
}

// ---------------- zero agg ----------------
__global__ __launch_bounds__(256) void zero_kernel(float4* __restrict__ p) {
  p[(size_t)blockIdx.x * 256 + threadIdx.x] = make_float4(0.f, 0.f, 0.f, 0.f);
}

// ---------------- encoder: h = seq2(enc, [off, x0, x1, x2']) -> f16 ----------------
__global__ __launch_bounds__(256) void encoder_kernel(
    const float* __restrict__ x, const float* __restrict__ W1,
    const float* __restrict__ B1, const float* __restrict__ W2,
    const float* __restrict__ B2, _Float16* __restrict__ hout, int root) {
  __shared__ float sW1[4 * 64], sB1s[64], sW2[64 * 64], sB2s[64];
  for (int t = threadIdx.x; t < 4 * 64; t += 256) sW1[t] = W1[t];
  for (int t = threadIdx.x; t < 64 * 64; t += 256) sW2[t] = W2[t];
  if (threadIdx.x < 64) { sB1s[threadIdx.x] = B1[threadIdx.x]; sB2s[threadIdx.x] = B2[threadIdx.x]; }
  __syncthreads();

  const int n = blockIdx.x * 256 + threadIdx.x;   // NN == 256 * gridDim
  const int b = n / NODES, loc = n - b * NODES;
  const float off = x[(size_t)(b * NODES + root) * 3 + 0];
  const float f0 = off;
  const float f1 = x[(size_t)n * 3 + 0];
  const float f2 = x[(size_t)n * 3 + 1];
  const float f3 = (loc == root) ? 1.0f : x[(size_t)n * 3 + 2];

  float h1[64];
#pragma unroll
  for (int j = 0; j < 64; ++j) {
    float v = sB1s[j] + f0 * sW1[j] + f1 * sW1[64 + j] + f2 * sW1[128 + j] + f3 * sW1[192 + j];
    h1[j] = v > 0.f ? v : 0.f;
  }
  for (int j = 0; j < 64; ++j) {
    float v = sB2s[j];
#pragma unroll 8
    for (int k = 0; k < 64; ++k) v += h1[k] * sW2[k * 64 + j];
    hout[(size_t)n * 64 + j] = (_Float16)v;
  }
}

// ---------------- conv edge: msg = seq3(mlp1, [h[src], eattr]); agg[dst] += msg ----------------
__global__ __launch_bounds__(256) void conv_edge_kernel(
    const long long* __restrict__ eidx,    // [2][E]
    const float* __restrict__ eattr,       // [E][8]
    const _Float16* __restrict__ h,        // [N][64]
    const float* __restrict__ W1, const float* __restrict__ B1,  // 72x64
    const float* __restrict__ W2, const float* __restrict__ B2,  // 64x64
    const float* __restrict__ W3, const float* __restrict__ B3,  // 64x64
    float* __restrict__ agg) {
  __shared__ __attribute__((aligned(16))) _Float16 sX0[8][16 * 96];
  __shared__ __attribute__((aligned(16))) _Float16 sX1[8][16 * 64];
  __shared__ __attribute__((aligned(16))) _Float16 sX2[8][16 * 64];
  __shared__ __attribute__((aligned(16))) _Float16 sW1t[64 * 96];  // [n][k], k padded 72->96
  __shared__ __attribute__((aligned(16))) _Float16 sW2t[64 * 64];
  __shared__ __attribute__((aligned(16))) _Float16 sW3t[64 * 64];
  __shared__ float sB1[64], sB2[64], sB3[64];

  const int tid = threadIdx.x, wave = tid >> 5, lane = tid & 31;
  const int mlo = lane & 15, hi = lane >> 4;

  // stage weights transposed + K-padded into LDS (L2-resident source); amortized over ETILES
  for (int t = tid; t < 64 * 96; t += 256) {
    const int n = t / 96, k = t - n * 96;
    sW1t[t] = (k < 72) ? (_Float16)W1[k * 64 + n] : (_Float16)0.f;
  }
  for (int t = tid; t < 64 * 64; t += 256) {
    const int n = t >> 6, k = t & 63;
    sW2t[t] = (_Float16)W2[k * 64 + n];
    sW3t[t] = (_Float16)W3[k * 64 + n];
  }
  if (tid < 64) { sB1[tid] = B1[tid]; sB2[tid] = B2[tid]; sB3[tid] = B3[tid]; }

  // zero-pad A columns 72..95 once (gather only overwrites cols 0..71)
  for (int t = lane; t < 16 * 6; t += 32) {
    const int row = t / 6, ch = t - row * 6;
    *(unsigned long long*)&sX0[wave][row * 96 + 72 + ch * 4] = 0ull;
  }
  __syncthreads();

  const long long eBase = (long long)blockIdx.x * (128 * ETILES) + wave * 16;
  const int row4 = lane >> 3, chunk = lane & 7;  // 4 rows x 8 x 16B per async iter

  for (int t = 0; t < ETILES; ++t) {
    const long long e0 = eBase + t * 128;
    if (t + 1 < ETILES) __builtin_prefetch(eattr + (e0 + 128) * 8, 0, 0);

    // per-wave edge indices (lanes 0..15), broadcast via wave32 shuffles
    int sIdx = 0, dIdx = 0;
    if (lane < 16) {
      sIdx = (int)eidx[e0 + lane];
      dIdx = (int)eidx[(long long)EE + e0 + lane];
    }

    // async gather: 16 rows x 128B of h[src] -> sX0 (no VGPR round-trip)
#pragma unroll
    for (int it = 0; it < 4; ++it) {
      const int row = it * 4 + row4;
      const int src = __shfl(sIdx, row);
      async_gather_b128(lds_offset(&sX0[wave][row * 96 + chunk * 8]),
                        h + (size_t)src * 64 + chunk * 8);
    }

    // edge_attr: 2 lanes per edge, float4 load + cvt + b64 LDS store
    {
      const int er = lane >> 1, part = lane & 1;
      const float4 a4 = *(const float4*)(eattr + (e0 + er) * 8 + part * 4);
      v4h hv = {(_Float16)a4.x, (_Float16)a4.y, (_Float16)a4.z, (_Float16)a4.w};
      *(v4h*)&sX0[wave][er * 96 + 64 + part * 4] = hv;
    }
    wait_asynccnt0();

    stage_relu<3>(&sX0[wave][0], 96, sW1t, sB1, &sX1[wave][0], lane);
    stage_relu<2>(&sX1[wave][0], 64, sW2t, sB2, &sX2[wave][0], lane);

    // final layer (no relu) + atomic scatter into agg[dst]
    const v16h a0 = load_a_frag(&sX2[wave][0], 64, mlo, hi, 0);
    const v16h a1 = load_a_frag(&sX2[wave][0], 64, mlo, hi, 32);
    int dn[8];
#pragma unroll
    for (int r = 0; r < 8; ++r) dn[r] = __shfl(dIdx, r + 8 * hi);
#pragma unroll
    for (int nt = 0; nt < 4; ++nt) {
      v8f c = {};
      c = wmma_f16(a0, load_b_frag(sW3t, 64, mlo, hi, 0, nt * 16), c);
      c = wmma_f16(a1, load_b_frag(sW3t, 64, mlo, hi, 32, nt * 16), c);
      const float bn = sB3[nt * 16 + mlo];
#pragma unroll
      for (int r = 0; r < 8; ++r)
        atomicAdd(agg + (size_t)dn[r] * 64 + nt * 16 + mlo, c[r] + bn);
    }
  }
}

// ---------------- conv node: h = seq3(mlp2, [h, agg]) (in place) ----------------
__global__ __launch_bounds__(256) void conv_node_kernel(
    _Float16* __restrict__ h,              // [N][64] in/out
    const float* __restrict__ agg,         // [N][64]
    const float* __restrict__ W1, const float* __restrict__ B1,  // 128x64
    const float* __restrict__ W2, const float* __restrict__ B2,  // 64x64
    const float* __restrict__ W3, const float* __restrict__ B3)  // 64x64
{
  __shared__ __attribute__((aligned(16))) _Float16 sX0[8][16 * 128];
  __shared__ __attribute__((aligned(16))) _Float16 sX1[8][16 * 64];
  __shared__ __attribute__((aligned(16))) _Float16 sX2[8][16 * 64];
  __shared__ __attribute__((aligned(16))) _Float16 sW1t[64 * 128];
  __shared__ __attribute__((aligned(16))) _Float16 sW2t[64 * 64];
  __shared__ __attribute__((aligned(16))) _Float16 sW3t[64 * 64];
  __shared__ float sB1[64], sB2[64], sB3[64];

  const int tid = threadIdx.x, wave = tid >> 5, lane = tid & 31;
  const int mlo = lane & 15, hi = lane >> 4;

  for (int t = tid; t < 64 * 128; t += 256) {
    const int n = t >> 7, k = t & 127;
    sW1t[t] = (_Float16)W1[k * 64 + n];
  }
  for (int t = tid; t < 64 * 64; t += 256) {
    const int n = t >> 6, k = t & 63;
    sW2t[t] = (_Float16)W2[k * 64 + n];
    sW3t[t] = (_Float16)W3[k * 64 + n];
  }
  if (tid < 64) { sB1[tid] = B1[tid]; sB2[tid] = B2[tid]; sB3[tid] = B3[tid]; }

  const int node0 = blockIdx.x * 128 + wave * 16;

  // async gather: own 16 contiguous h rows -> sX0 cols 0..63
  {
    const int row4 = lane >> 3, chunk = lane & 7;
#pragma unroll
    for (int it = 0; it < 4; ++it) {
      const int row = it * 4 + row4;
      async_gather_b128(lds_offset(&sX0[wave][row * 128 + chunk * 8]),
                        h + (size_t)(node0 + row) * 64 + chunk * 8);
    }
  }
  // agg (f32, contiguous 1024 floats) -> f16 cols 64..127
#pragma unroll
  for (int it = 0; it < 8; ++it) {
    const int f4i = it * 32 + lane;             // 0..255 float4s
    const float4 a4 = *(const float4*)(agg + (size_t)node0 * 64 + (size_t)f4i * 4);
    const int g0 = f4i * 4, row = g0 >> 6, col = g0 & 63;
    v4h hv = {(_Float16)a4.x, (_Float16)a4.y, (_Float16)a4.z, (_Float16)a4.w};
    *(v4h*)&sX0[wave][row * 128 + 64 + col] = hv;
  }
  wait_asynccnt0();
  __syncthreads();

  stage_relu<4>(&sX0[wave][0], 128, sW1t, sB1, &sX1[wave][0], lane);
  stage_relu<2>(&sX1[wave][0], 64,  sW2t, sB2, &sX2[wave][0], lane);

  const v16h a0 = load_a_frag(&sX2[wave][0], 64, mlo, hi, 0);
  const v16h a1 = load_a_frag(&sX2[wave][0], 64, mlo, hi, 32);
#pragma unroll
  for (int nt = 0; nt < 4; ++nt) {
    v8f c = {};
    c = wmma_f16(a0, load_b_frag(sW3t, 64, mlo, hi, 0, nt * 16), c);
    c = wmma_f16(a1, load_b_frag(sW3t, 64, mlo, hi, 32, nt * 16), c);
    const float bn = sB3[nt * 16 + mlo];
#pragma unroll
    for (int r = 0; r < 8; ++r)
      h[(size_t)(node0 + r + 8 * hi) * 64 + nt * 16 + mlo] = (_Float16)(c[r] + bn);
  }
}

// ---------------- decoder ----------------
__global__ __launch_bounds__(256) void dec1_kernel(
    const float* __restrict__ x, const _Float16* __restrict__ h,
    const float* __restrict__ W1, const float* __restrict__ B1,
    float* __restrict__ out, int root) {
  __shared__ float sW[65 * 256];
  __shared__ float sF[65];
  const int b = blockIdx.x;
  const int node = b * NODES + root;
  for (int t = threadIdx.x; t < 65 * 256; t += 256) sW[t] = W1[t];
  if (threadIdx.x == 0) sF[0] = x[(size_t)node * 3 + 0];
  else if (threadIdx.x <= 64) sF[threadIdx.x] = (float)h[(size_t)node * 64 + threadIdx.x - 1];
  __syncthreads();
  const int col = threadIdx.x;
  float v = B1[col];
#pragma unroll 5
  for (int k = 0; k < 65; ++k) v += sF[k] * sW[k * 256 + col];
  out[(size_t)b * 256 + col] = v > 0.f ? v : 0.f;
}

__global__ __launch_bounds__(64) void dec2_kernel(
    const float* __restrict__ in, const float* __restrict__ W2,
    const float* __restrict__ B2, float* __restrict__ out) {
  __shared__ float sW[256 * 64];
  __shared__ float sIn[256];
  for (int t = threadIdx.x; t < 256 * 64; t += 64) sW[t] = W2[t];
  for (int t = threadIdx.x; t < 256; t += 64) sIn[t] = in[(size_t)blockIdx.x * 256 + t];
  __syncthreads();
  float v = B2[threadIdx.x];
#pragma unroll 8
  for (int k = 0; k < 256; ++k) v += sIn[k] * sW[k * 64 + threadIdx.x];
  out[(size_t)blockIdx.x * 64 + threadIdx.x] = v > 0.f ? v : 0.f;
}

__global__ __launch_bounds__(256) void dec3_kernel(
    const float* __restrict__ in, const float* __restrict__ W3,
    const float* __restrict__ B3, float* __restrict__ out, int i) {
  const int row = blockIdx.x * 256 + threadIdx.x;  // grid 4 x 256 == 1024
  float v = B3[0];
#pragma unroll 8
  for (int k = 0; k < 64; ++k) v += in[(size_t)row * 64 + k] * W3[k];
  out[(size_t)row * DIMQ + i] = v;
}

// ---------------- host launcher ----------------
// d_in order: x, edge_index(i64), edge_attr, bz, dimq, omega_p, batch(i64),
// then params leaves in jax pytree order (dict keys sorted: convs < decoder < encoder):
//   convs[c]: mlp1 {W 72x64, b64, W 64x64, b64, W 64x64, b64},
//             mlp2 {W 128x64, b64, W 64x64, b64, W 64x64, b64}   (12 leaves x 5)
//   decoder:  W 65x256, b256, W 256x64, b64, W 64x1, b1           (6 leaves)
//   encoder:  W 4x64, b64, W 64x64, b64                           (4 leaves)
extern "C" void kernel_launch(void* const* d_in, const int* in_sizes, int n_in,
                              void* d_out, int out_size, void* d_ws, size_t ws_size,
                              hipStream_t stream) {
  (void)in_sizes; (void)n_in; (void)out_size; (void)ws_size;

  const float*     x     = (const float*)d_in[0];
  const long long* eidx  = (const long long*)d_in[1];
  const float*     eattr = (const float*)d_in[2];
  const int P = 7;  // first params leaf
  auto F = [&](int i) { return (const float*)d_in[i]; };

  const float *decW1 = F(P + 60), *decB1 = F(P + 61), *decW2 = F(P + 62),
              *decB2 = F(P + 63), *decW3 = F(P + 64), *decB3 = F(P + 65);
  const float *encW1 = F(P + 66), *encB1 = F(P + 67), *encW2 = F(P + 68),
              *encB2 = F(P + 69);

  // workspace layout
  char* ws = (char*)d_ws;
  size_t o = 0;
  _Float16* hf16 = (_Float16*)(ws + o);
  o += (((size_t)NN * 64 * sizeof(_Float16)) + 255) & ~(size_t)255;
  float* agg = (float*)(ws + o);
  o += (((size_t)NN * 64 * sizeof(float)) + 255) & ~(size_t)255;
  float* d1buf = (float*)(ws + o);
  o += (size_t)1024 * 256 * sizeof(float);
  float* d2buf = (float*)(ws + o);

  for (int i = 0; i < DIMQ; ++i) {
    const int root = BZ * DIMQ + i;
    encoder_kernel<<<NN / 256, 256, 0, stream>>>(x, encW1, encB1, encW2, encB2,
                                                 hf16, root);
    for (int c = 0; c < 5; ++c) {
      const int b = P + c * 12;
      zero_kernel<<<(NN * 64) / 1024, 256, 0, stream>>>((float4*)agg);
      conv_edge_kernel<<<EE / (128 * ETILES), 256, 0, stream>>>(
          eidx, eattr, hf16, F(b + 0), F(b + 1), F(b + 2), F(b + 3), F(b + 4),
          F(b + 5), agg);
      conv_node_kernel<<<NN / 128, 256, 0, stream>>>(
          hf16, agg, F(b + 6), F(b + 7), F(b + 8), F(b + 9), F(b + 10),
          F(b + 11));
    }
    dec1_kernel<<<1024, 256, 0, stream>>>(x, hf16, decW1, decB1, d1buf, root);
    dec2_kernel<<<1024, 64, 0, stream>>>(d1buf, decW2, decB2, d2buf);
    dec3_kernel<<<4, 256, 0, stream>>>(d2buf, decW3, decB3, (float*)d_out, i);
  }
}
